// FastAttention_88811333747053
// MI455X (gfx1250) — compile-verified, matrix operand order checked by reading, not run on hardware
//
#include <hip/hip_runtime.h>
#include <hip/hip_bf16.h>

// Performer (FAVOR+) linear attention for MI455X / gfx1250.
// All four GEMMs use v_wmma_f32_16x16x32_bf16 (wave32 WMMA).
// Feature tensors never hit HBM; proj is pre-converted to bf16 once so all
// B-fragments are raw b128 loads (the 16-bit frag layout is two contiguous
// 8-element chunks per lane: K = [c, c+8) U [c+16, c+24), c = 8h + 32kc).

typedef __attribute__((ext_vector_type(16))) __bf16 v16bf;
typedef __attribute__((ext_vector_type(8)))  __bf16 v8bf;
typedef __attribute__((ext_vector_type(8)))  float  v8f;

#define BH    16      // B*H
#define SEQ   8192    // N
#define DDIM  64      // d
#define FDIM  256     // nb_features
#define EDIM  64      // v head dim

#define DATA_NORM 0.35355339059327379f  // 64^-0.25
#define DIAG_SCL  0.0625f               // 0.5 * 64^-0.5
#define RATIO     0.0625f               // 256^-0.5
#define KEPS      1e-4f

// ---- order-preserving float <-> uint key for atomicMax over floats ----
__device__ __forceinline__ unsigned fkey(float x) {
  unsigned u = __float_as_uint(x);
  return (u & 0x80000000u) ? ~u : (u | 0x80000000u);
}
__device__ __forceinline__ float fdecode(unsigned kk) {
  unsigned u = (kk & 0x80000000u) ? (kk ^ 0x80000000u) : ~kk;
  return __uint_as_float(u);
}

// Load a 16-bit A/B WMMA fragment from a row-major bf16 row (stride in
// 8-element units): two contiguous 16B chunks at chunk indices h+4kc, h+4kc+2.
__device__ __forceinline__ v16bf load_frag_bf(const __bf16* rowBase, int h, int kc) {
  const v8bf* p = (const v8bf*)rowBase;
  v8bf lo = p[h + 4 * kc];
  v8bf hi = p[h + 4 * kc + 2];
  return __builtin_shufflevector(lo, hi, 0, 1, 2, 3, 4, 5, 6, 7,
                                 8, 9, 10, 11, 12, 13, 14, 15);
}

// Build an A fragment from an fp32 row (scaled), accumulating raw sum(x*x).
__device__ __forceinline__ v16bf make_afrag(const float* __restrict__ rowPtr,
                                            int h, int kc, float& ss) {
  const int c = 8 * h + 32 * kc;
  float4 a0 = *(const float4*)(rowPtr + c);
  float4 a1 = *(const float4*)(rowPtr + c + 4);
  float4 b0 = *(const float4*)(rowPtr + c + 16);
  float4 b1 = *(const float4*)(rowPtr + c + 20);
  ss += a0.x * a0.x + a0.y * a0.y + a0.z * a0.z + a0.w * a0.w +
        a1.x * a1.x + a1.y * a1.y + a1.z * a1.z + a1.w * a1.w +
        b0.x * b0.x + b0.y * b0.y + b0.z * b0.z + b0.w * b0.w +
        b1.x * b1.x + b1.y * b1.y + b1.z * b1.z + b1.w * b1.w;
  v16bf a;
  a[0]  = (__bf16)(a0.x * DATA_NORM); a[1]  = (__bf16)(a0.y * DATA_NORM);
  a[2]  = (__bf16)(a0.z * DATA_NORM); a[3]  = (__bf16)(a0.w * DATA_NORM);
  a[4]  = (__bf16)(a1.x * DATA_NORM); a[5]  = (__bf16)(a1.y * DATA_NORM);
  a[6]  = (__bf16)(a1.z * DATA_NORM); a[7]  = (__bf16)(a1.w * DATA_NORM);
  a[8]  = (__bf16)(b0.x * DATA_NORM); a[9]  = (__bf16)(b0.y * DATA_NORM);
  a[10] = (__bf16)(b0.z * DATA_NORM); a[11] = (__bf16)(b0.w * DATA_NORM);
  a[12] = (__bf16)(b1.x * DATA_NORM); a[13] = (__bf16)(b1.y * DATA_NORM);
  a[14] = (__bf16)(b1.z * DATA_NORM); a[15] = (__bf16)(b1.w * DATA_NORM);
  return a;
}

// Per-wave feature GEMM: acc[nt] = (DATA_NORM*data[16x64]) @ projBf[256x64]^T
// tile-of-16 (16 rows x 256 features, 32 wmma). sumsq = raw per-row sum(x*x)
// for row = lane&15 (replicated across both lane halves).
__device__ __forceinline__ void feat_gemm(const float* __restrict__ dataRows,
                                          const __bf16* __restrict__ projBf,
                                          v8f acc[16], float& sumsq) {
  const int lane = threadIdx.x & 31;
  const int m = lane & 15;
  const int h = lane >> 4;
  float ss = 0.f;
  v16bf afrag[2];
#pragma unroll
  for (int kc = 0; kc < 2; ++kc)
    afrag[kc] = make_afrag(dataRows + m * DDIM, h, kc, ss);
  sumsq = ss + __shfl_xor(ss, 16, 32);  // lane pair covers complementary K halves
#pragma unroll
  for (int nt = 0; nt < 16; ++nt) {
    v8f c = {0.f, 0.f, 0.f, 0.f, 0.f, 0.f, 0.f, 0.f};
#pragma unroll
    for (int kc = 0; kc < 2; ++kc) {
      v16bf b = load_frag_bf(projBf + (nt * 16 + m) * DDIM, h, kc);
      c = __builtin_amdgcn_wmma_f32_16x16x32_bf16(false, afrag[kc], false, b,
                                                  (short)0, c, false, false);
    }
    acc[nt] = c;
  }
}

// ---------------- Pass 0: convert projection to bf16 once ----------------
__global__ void __launch_bounds__(256)
proj_prep_kernel(const float* __restrict__ proj, __bf16* __restrict__ projBf) {
  const int idx = blockIdx.x * 256 + threadIdx.x;
  if (idx < FDIM * DDIM) projBf[idx] = (__bf16)proj[idx];
}

// ---------------- Pass 1: global max over k feature logits ----------------
__global__ void __launch_bounds__(128)
kmax_kernel(const float* __restrict__ kmat, const __bf16* __restrict__ projBf,
            unsigned* __restrict__ maxkey) {
  const int bh = blockIdx.y;
  const int wave = threadIdx.x >> 5;
  const int lane = threadIdx.x & 31;
  const long rowBase = (long)bh * SEQ + (long)blockIdx.x * 64 + wave * 16;
  v8f acc[16];
  float ss;
  feat_gemm(kmat + rowBase * DDIM, projBf, acc, ss);
  float mx = -3.402823466e38f;
#pragma unroll
  for (int nt = 0; nt < 16; ++nt)
#pragma unroll
    for (int r = 0; r < 8; ++r) mx = fmaxf(mx, acc[nt][r]);
#pragma unroll
  for (int mask = 16; mask >= 1; mask >>= 1)
    mx = fmaxf(mx, __shfl_xor(mx, mask, 32));
  if (lane == 0) atomicMax(maxkey, fkey(mx));
}

// ------- Pass 2: kp = feature map of k; accumulate k_sum and context -------
__global__ void __launch_bounds__(128)
kctx_kernel(const float* __restrict__ kmat, const float* __restrict__ vmat,
            const __bf16* __restrict__ projBf, const unsigned* __restrict__ maxkey,
            float* __restrict__ ksum, float* __restrict__ ctx) {
  __shared__ __bf16 sKpT[FDIM * 64];  // [f][row_local]  32 KB (frag rows = f)
  __shared__ __bf16 sVT[EDIM * 64];   // [e][row_local]   8 KB (frag rows = e)
  const int bh = blockIdx.y;
  const int wave = threadIdx.x >> 5;
  const int lane = threadIdx.x & 31;
  const int m = lane & 15, h = lane >> 4;
  const long rows0 = (long)bh * SEQ + (long)blockIdx.x * 64;

  // stage v (64 rows x 64) transposed into LDS as bf16 (coalesced global reads)
  for (int i = threadIdx.x; i < 64 * EDIM; i += 128) {
    int row = i >> 6, e = i & 63;
    sVT[e * 64 + row] = (__bf16)vmat[(rows0 + row) * EDIM + e];
  }

  v8f acc[16];
  float ss;
  feat_gemm(kmat + (rows0 + wave * 16) * DDIM, projBf, acc, ss);
  const float stab = fdecode(*maxkey);

  float diag[8];
#pragma unroll
  for (int r = 0; r < 8; ++r)
    diag[r] = DIAG_SCL * __shfl(ss, r + 8 * h, 32);  // C row = r + 8h

#pragma unroll
  for (int nt = 0; nt < 16; ++nt) {
    float colsum = 0.f;
#pragma unroll
    for (int r = 0; r < 8; ++r) {
      float kp = RATIO * (__expf(acc[nt][r] - diag[r] - stab) + KEPS);
      colsum += kp;
      sKpT[(nt * 16 + m) * 64 + wave * 16 + r + 8 * h] = (__bf16)kp;
    }
    colsum += __shfl_xor(colsum, 16, 32);
    if (h == 0) atomicAdd(&ksum[bh * FDIM + nt * 16 + m], colsum);
  }
  __syncthreads();

  // context[256x64] += kp^T[256 x 64rows] @ v[64rows x 64]; wave owns 4 f-tiles.
#pragma unroll
  for (int fti = 0; fti < 4; ++fti) {
    const int ft = wave * 4 + fti;
    v16bf a[2];
#pragma unroll
    for (int kc = 0; kc < 2; ++kc)
      a[kc] = load_frag_bf(sKpT + (ft * 16 + m) * 64, h, kc);
#pragma unroll
    for (int et = 0; et < 4; ++et) {
      v8f c = {0.f, 0.f, 0.f, 0.f, 0.f, 0.f, 0.f, 0.f};
#pragma unroll
      for (int kc = 0; kc < 2; ++kc) {
        v16bf b = load_frag_bf(sVT + (et * 16 + m) * 64, h, kc);
        c = __builtin_amdgcn_wmma_f32_16x16x32_bf16(false, a[kc], false, b,
                                                    (short)0, c, false, false);
      }
#pragma unroll
      for (int r = 0; r < 8; ++r)
        atomicAdd(&ctx[((long)bh * FDIM + ft * 16 + r + 8 * h) * EDIM + et * 16 + m],
                  c[r]);
    }
  }
}

// ------ Pass 3: qp feature map fused with out = (qp @ context) * d^-1 ------
__global__ void __launch_bounds__(128)
qout_kernel(const float* __restrict__ qmat, const __bf16* __restrict__ projBf,
            const float* __restrict__ ksum, const float* __restrict__ ctx,
            float* __restrict__ out) {
  __shared__ __bf16 sQp[4 * 16 * FDIM];   // [row_local][f] 32 KB
  __shared__ __bf16 sCtxT[EDIM * FDIM];   // [e][f]         32 KB
  const int bh = blockIdx.y;
  const int wave = threadIdx.x >> 5;
  const int lane = threadIdx.x & 31;
  const int m = lane & 15, h = lane >> 4;
  const long rows0 = (long)bh * SEQ + (long)blockIdx.x * 64;

  // stage context transposed into LDS as bf16
  for (int i = threadIdx.x; i < FDIM * EDIM; i += 128) {
    int f = i >> 6, e = i & 63;
    sCtxT[e * FDIM + f] = (__bf16)ctx[((long)bh * FDIM + f) * EDIM + e];
  }

  float ks[16];
#pragma unroll
  for (int nt = 0; nt < 16; ++nt) ks[nt] = ksum[bh * FDIM + nt * 16 + m];

  v8f acc[16];
  float ss;
  feat_gemm(qmat + (rows0 + wave * 16) * DDIM, projBf, acc, ss);

  float diag[8], stab[8], den[8];
#pragma unroll
  for (int r = 0; r < 8; ++r) {
    diag[r] = DIAG_SCL * __shfl(ss, r + 8 * h, 32);
    float mx = -3.402823466e38f;
#pragma unroll
    for (int nt = 0; nt < 16; ++nt) mx = fmaxf(mx, acc[nt][r]);
#pragma unroll
    for (int mask = 8; mask >= 1; mask >>= 1)
      mx = fmaxf(mx, __shfl_xor(mx, mask, 32));  // row max across half-wave lanes
    stab[r] = mx;
    den[r] = 0.f;
  }
#pragma unroll
  for (int nt = 0; nt < 16; ++nt) {
#pragma unroll
    for (int r = 0; r < 8; ++r) {
      float qp = RATIO * (__expf(acc[nt][r] - diag[r] - stab[r]) + KEPS);
      den[r] += qp * ks[nt];
      sQp[(wave * 16 + r + 8 * h) * FDIM + nt * 16 + m] = (__bf16)qp;
    }
  }
  float dinv[8];
#pragma unroll
  for (int r = 0; r < 8; ++r) {
    float d = den[r];
#pragma unroll
    for (int mask = 8; mask >= 1; mask >>= 1) d += __shfl_xor(d, mask, 32);
    dinv[r] = 1.0f / d;
  }
  __syncthreads();

  // out[16x64] = qp[16x256] @ ctx[256x64]; A frags shared by all 4 out tiles.
  v16bf a[8];
#pragma unroll
  for (int kc = 0; kc < 8; ++kc)
    a[kc] = load_frag_bf(sQp + (wave * 16 + m) * FDIM, h, kc);
#pragma unroll
  for (int et = 0; et < 4; ++et) {
    v8f c = {0.f, 0.f, 0.f, 0.f, 0.f, 0.f, 0.f, 0.f};
#pragma unroll
    for (int kc = 0; kc < 8; ++kc) {
      v16bf b = load_frag_bf(sCtxT + (et * 16 + m) * FDIM, h, kc);
      c = __builtin_amdgcn_wmma_f32_16x16x32_bf16(false, a[kc], false, b,
                                                  (short)0, c, false, false);
    }
#pragma unroll
    for (int r = 0; r < 8; ++r)
      out[(rows0 + wave * 16 + r + 8 * h) * EDIM + et * 16 + m] = c[r] * dinv[r];
  }
}

extern "C" void kernel_launch(void* const* d_in, const int* in_sizes, int n_in,
                              void* d_out, int out_size, void* d_ws, size_t ws_size,
                              hipStream_t stream) {
  (void)in_sizes; (void)n_in; (void)out_size; (void)ws_size;
  const float* q    = (const float*)d_in[0];
  const float* k    = (const float*)d_in[1];
  const float* v    = (const float*)d_in[2];
  const float* proj = (const float*)d_in[3];
  float* out = (float*)d_out;

  float* ws = (float*)d_ws;
  unsigned* maxkey = (unsigned*)ws;                      // [0]
  float* ksum = ws + 64;                                 // [BH*256]
  float* ctx  = ws + 64 + BH * FDIM;                     // [BH*256*64]
  __bf16* projBf = (__bf16*)(ws + 64 + BH * FDIM + (size_t)BH * FDIM * EDIM);

  const size_t clearBytes =
      (64 + (size_t)BH * FDIM + (size_t)BH * FDIM * EDIM) * sizeof(float);
  hipMemsetAsync(d_ws, 0, clearBytes, stream);  // zero accumulators; key 0 == -inf

  dim3 grid(SEQ / 64, BH);
  proj_prep_kernel<<<(FDIM * DDIM + 255) / 256, 256, 0, stream>>>(proj, projBf);
  kmax_kernel<<<grid, 128, 0, stream>>>(k, projBf, maxkey);
  kctx_kernel<<<grid, 128, 0, stream>>>(k, v, projBf, maxkey, ksum, ctx);
  qout_kernel<<<grid, 128, 0, stream>>>(q, projBf, ksum, ctx, out);
}
